// MoELayer_39195871543600
// MI455X (gfx1250) — matrix operand order repaired and checked by reading, not compiled, not run on hardware
//
#include <hip/hip_runtime.h>
#include <hip/hip_bf16.h>
#include <math.h>

// ---------------------------------------------------------------------------
// MoE layer for MI455X (gfx1250): top-2 routed expert FFN with bf16 WMMA.
// Round 4: revert register-prefetch pipelining (it spilled to scratch);
// staging is a self-contained global->LDS copy phase again. Keep padded sA
// (conflict-free A-fragment ds_load_b128) and hoisted staging pointers; add
// zero-register global_prefetch_b8 of the next K-chunk during the WMMA phase.
// ---------------------------------------------------------------------------

typedef __attribute__((ext_vector_type(16))) __bf16 v16bf;
typedef __attribute__((ext_vector_type(8)))  __bf16 v8bf;
typedef __attribute__((ext_vector_type(8)))  float  v8f;

#define N_TOK  16384
#define DMODEL 1024
#define NEXP   8
#define HDIM   2048
#define TOPK   2
// padded slot capacity: every expert region rounded up to 64 rows
#define SLOT_CAP (N_TOK * TOPK + NEXP * 64)   // 33280

// float -> bf16 round-to-nearest-even (as raw ushort)
__device__ __forceinline__ unsigned short f2bf(float f) {
    unsigned u = __float_as_uint(f);
    u += 0x7FFFu + ((u >> 16) & 1u);
    return (unsigned short)(u >> 16);
}

// Branchless GELU (exact-erf form, Abramowitz-Stegun 7.1.26, |err| < 1.5e-7 —
// far below the bf16 matmul noise floor). One v_exp + v_rcp + ~10 FMAs.
__device__ __forceinline__ float gelu_f(float v) {
    float s = v * 0.70710678118654752f;
    float a = fabsf(s);
    float t = __builtin_amdgcn_rcpf(1.0f + 0.3275911f * a);
    float p = t * (0.254829592f +
              t * (-0.284496736f +
              t * (1.421413741f +
              t * (-1.453152027f +
              t * 1.061405429f))));
    float erfa = 1.0f - p * __expf(-a * a);
    float erfv = copysignf(erfa, s);
    return 0.5f * v * (1.0f + erfv);
}

// ---------------------------------------------------------------------------
__global__ void cvt_bf16_kernel(const float* __restrict__ src,
                                unsigned short* __restrict__ dst, int n) {
    int i = blockIdx.x * blockDim.x + threadIdx.x;
    int stride = gridDim.x * blockDim.x;
    for (; i < n; i += stride) dst[i] = f2bf(src[i]);
}

// convert + transpose: src [E][R][C] fp32 -> dst [E][C][R] bf16.
// 32x32 LDS tile so both the global read and the global write are coalesced.
__global__ void cvtT_kernel(const float* __restrict__ src,
                            unsigned short* __restrict__ dst, int R, int C) {
    __shared__ unsigned short tile[32][33];
    const int e = blockIdx.z;
    const float* s = src + (size_t)e * R * C;
    unsigned short* d = dst + (size_t)e * R * C;
    const int c0 = blockIdx.x * 32, r0 = blockIdx.y * 32;
#pragma unroll
    for (int k = 0; k < 4; k++) {
        int r = threadIdx.y + k * 8;
        tile[r][threadIdx.x] = f2bf(s[(size_t)(r0 + r) * C + c0 + threadIdx.x]);
    }
    __syncthreads();
#pragma unroll
    for (int k = 0; k < 4; k++) {
        int c = threadIdx.y + k * 8;
        d[(size_t)(c0 + c) * R + r0 + threadIdx.x] = tile[threadIdx.x][c];
    }
}

// out = x (residual), map = -1, counts/cursors = 0  (re-done every launch)
__global__ void init_kernel(const float* __restrict__ x, float* __restrict__ out,
                            int* __restrict__ map, int* __restrict__ counts,
                            int* __restrict__ cursor) {
    int i = blockIdx.x * blockDim.x + threadIdx.x;
    int stride = gridDim.x * blockDim.x;
    for (int j = i; j < N_TOK * DMODEL; j += stride) out[j] = x[j];
    for (int j = i; j < SLOT_CAP; j += stride) map[j] = -1;
    if (i < NEXP) { counts[i] = 0; cursor[i] = 0; }
}

// ---------------------------------------------------------------------------
// Gating: one wave32 per token. logits = x@Wg + bg; top-2; 2-way softmax
// (identical to softmax -> top_k -> renormalize, since Z cancels).
__global__ void gate_kernel(const float* __restrict__ x, const float* __restrict__ Wg,
                            const float* __restrict__ bg, int* __restrict__ topi,
                            float* __restrict__ topw, int* __restrict__ counts) {
    int gwave = (blockIdx.x * blockDim.x + threadIdx.x) >> 5;
    int lane  = threadIdx.x & 31;
    if (gwave >= N_TOK) return;
    const float* xr = x + (size_t)gwave * DMODEL;
    float acc[NEXP] = {};
    for (int d = lane; d < DMODEL; d += 32) {
        float xv = xr[d];
        const float* wr = Wg + d * NEXP;
#pragma unroll
        for (int e = 0; e < NEXP; e++) acc[e] += xv * wr[e];
    }
#pragma unroll
    for (int e = 0; e < NEXP; e++)
        for (int off = 16; off; off >>= 1) acc[e] += __shfl_xor(acc[e], off, 32);
    if (lane == 0) {
        float l[NEXP];
#pragma unroll
        for (int e = 0; e < NEXP; e++) l[e] = acc[e] + bg[e];
        int i0 = 0;
#pragma unroll
        for (int e = 1; e < NEXP; e++) if (l[e] > l[i0]) i0 = e;   // first-max tiebreak
        int i1 = -1;
#pragma unroll
        for (int e = 0; e < NEXP; e++)
            if (e != i0 && (i1 < 0 || l[e] > l[i1])) i1 = e;
        float w0 = 1.0f / (1.0f + __expf(l[i1] - l[i0]));  // renormalized 2-way softmax
        topi[2 * gwave]     = i0;  topi[2 * gwave + 1] = i1;
        topw[2 * gwave]     = w0;  topw[2 * gwave + 1] = 1.0f - w0;
        atomicAdd(&counts[i0], 1);
        atomicAdd(&counts[i1], 1);
    }
}

// exclusive scan of 64-rounded counts -> padded offsets (single thread, E=8)
__global__ void scan_kernel(const int* __restrict__ counts, int* __restrict__ offp) {
    if (blockIdx.x == 0 && threadIdx.x == 0) {
        int o = 0;
        for (int e = 0; e < NEXP; e++) {
            offp[e] = o;
            o += (counts[e] + 63) & ~63;
        }
        offp[NEXP] = o;
    }
}

__global__ void scatter_kernel(const int* __restrict__ topi, const float* __restrict__ topw,
                               const int* __restrict__ offp, int* __restrict__ cursor,
                               int* __restrict__ map, float* __restrict__ wgt) {
    int t = blockIdx.x * blockDim.x + threadIdx.x;
    if (t >= N_TOK) return;
#pragma unroll
    for (int k = 0; k < TOPK; k++) {
        int e = topi[2 * t + k];
        int pos = atomicAdd(&cursor[e], 1);
        int s = offp[e] + pos;
        map[s] = t;
        wgt[s] = topw[2 * t + k];
    }
}

// ---------------------------------------------------------------------------
// GEMM tiling shared by FC1/FC2: block tile 64(M) x 256(N), K-step 32,
// 8 waves in a 2(M) x 4(N) grid, each wave owns 32x64 = 2x4 WMMA tiles.
// Weights pre-transposed to N-major so both sA and sB staging are contiguous
// 16B copies. One designated thread per row issues global_prefetch_b8 for the
// next K-chunk during the WMMA phase (zero register cost, no counters).
//
// A fragment (16x32 bf16, per ISA): lane<16 -> row=lane, K{0..7},{16..23};
//                                   lane>=16 -> row=lane-16, K{8..15},{24..31}.
// B fragment (32x16 bf16): lane<16 -> col=lane, K{0..7},{16..23}; lane>=16 -> K{8..15},{24..31}.
// C tile (16x16 f32): element r -> row = r + (lane<16?0:8), col = lane&15.
// Both LDS tiles use a 40-ushort (80B) row stride: bank = row*20 mod 64 is a
// 4-spaced permutation over 16 lanes -> conflict-free b128 fragment loads.

__global__ __launch_bounds__(256)
void fc1_kernel(const unsigned short* __restrict__ xb,
                const unsigned short* __restrict__ w1t,   // [E][H][D] bf16
                const float* __restrict__ b1,
                const int* __restrict__ map,
                const int* __restrict__ offp,
                unsigned short* __restrict__ h1) {
    __shared__ unsigned short sA[64][40];     // M x K (padded)
    __shared__ unsigned short sB[256][40];    // N x K (padded)

    const int e     = blockIdx.z;
    const int base  = offp[e];
    const int cntp  = offp[e + 1] - base;
    const int mtile = blockIdx.x;
    if (mtile * 64 >= cntp) return;           // worst-case grid, device-side mask
    const int n0   = blockIdx.y * 256;
    const int tid  = threadIdx.x;
    const int lane = tid & 31;
    const int w    = tid >> 5;
    const int wm   = (w >> 2) * 32;
    const int wn   = (w & 3) * 64;
    const int lg   = lane >> 4;
    const int lm   = lane & 15;

    // staging jobs: A = 16B of one gathered row; B = 4x 16B of weight rows
    const int ar = tid >> 2, aq = tid & 3;
    int tok = map[base + mtile * 64 + ar];
    if (tok < 0) tok = 0;                     // padded row: harmless, masked in fc2
    const unsigned short* aSrc = xb + (size_t)tok * DMODEL + aq * 8;
    const unsigned short* bBase = w1t + (size_t)e * HDIM * DMODEL + (size_t)n0 * DMODEL;
    unsigned short* sAdst = &sA[ar][aq * 8];
    const unsigned short* bPtr[4];
    unsigned short* sBdst[4];
#pragma unroll
    for (int rep = 0; rep < 4; rep++) {
        int lin = tid + rep * 256;            // 0..1023 -> 256 rows x 4 quads
        int row = lin >> 2, q = lin & 3;
        bPtr[rep]  = bBase + (size_t)row * DMODEL + q * 8;
        sBdst[rep] = &sB[row][q * 8];
    }

    v8f acc[2][4] = {};

    for (int k0 = 0; k0 < DMODEL; k0 += 32) {
        __syncthreads();
        *(uint4*)sAdst = *(const uint4*)(aSrc + k0);
#pragma unroll
        for (int rep = 0; rep < 4; rep++)
            *(uint4*)sBdst[rep] = *(const uint4*)(bPtr[rep] + k0);
        __syncthreads();

        // pull next chunk's cachelines toward the WGP while WMMAs run
        if (k0 + 32 < DMODEL && aq == 0) {
            __builtin_prefetch(aSrc + k0 + 32, 0, 3);
#pragma unroll
            for (int rep = 0; rep < 4; rep++)
                __builtin_prefetch(bPtr[rep] + k0 + 32, 0, 3);
        }

        v16bf afrag[2], bfrag[4];
#pragma unroll
        for (int mi = 0; mi < 2; mi++) {
            int row = wm + mi * 16 + lm;
            v8bf a0 = *(const v8bf*)&sA[row][lg * 8];
            v8bf a1 = *(const v8bf*)&sA[row][16 + lg * 8];
            afrag[mi] = __builtin_shufflevector(a0, a1, 0,1,2,3,4,5,6,7,8,9,10,11,12,13,14,15);
        }
#pragma unroll
        for (int ni = 0; ni < 4; ni++) {
            int col = wn + ni * 16 + lm;
            v8bf b0 = *(const v8bf*)&sB[col][lg * 8];
            v8bf b1v = *(const v8bf*)&sB[col][16 + lg * 8];
            bfrag[ni] = __builtin_shufflevector(b0, b1v, 0,1,2,3,4,5,6,7,8,9,10,11,12,13,14,15);
        }
#pragma unroll
        for (int mi = 0; mi < 2; mi++)
#pragma unroll
            for (int ni = 0; ni < 4; ni++)
                acc[mi][ni] = __builtin_amdgcn_wmma_f32_16x16x32_bf16(
                    false, afrag[mi], false, bfrag[ni], (short)0, acc[mi][ni], false, false);
    }

    // epilogue: bias + branchless GELU -> bf16 H buffer
    const float* bias = b1 + (size_t)e * HDIM;
#pragma unroll
    for (int mi = 0; mi < 2; mi++)
#pragma unroll
        for (int ni = 0; ni < 4; ni++) {
            int col = n0 + wn + ni * 16 + lm;
            float bv = bias[col];
#pragma unroll
            for (int r = 0; r < 8; r++) {
                int row = wm + mi * 16 + r + lg * 8;
                int slotRow = base + mtile * 64 + row;
                h1[(size_t)slotRow * HDIM + col] = f2bf(gelu_f(acc[mi][ni][r] + bv));
            }
        }
}

__global__ __launch_bounds__(256)
void fc2_kernel(const unsigned short* __restrict__ h1,
                const unsigned short* __restrict__ w2t,   // [E][D][H] bf16
                const float* __restrict__ b2,
                const int* __restrict__ map,
                const int* __restrict__ offp,
                const float* __restrict__ wgt,
                float* __restrict__ out) {
    __shared__ unsigned short sA[64][40];
    __shared__ unsigned short sB[256][40];

    const int e     = blockIdx.z;
    const int base  = offp[e];
    const int cntp  = offp[e + 1] - base;
    const int mtile = blockIdx.x;
    if (mtile * 64 >= cntp) return;
    const int n0   = blockIdx.y * 256;
    const int tid  = threadIdx.x;
    const int lane = tid & 31;
    const int w    = tid >> 5;
    const int wm   = (w >> 2) * 32;
    const int wn   = (w & 3) * 64;
    const int lg   = lane >> 4;
    const int lm   = lane & 15;

    const int ar = tid >> 2, aq = tid & 3;
    const unsigned short* aSrc = h1 + (size_t)(base + mtile * 64 + ar) * HDIM + aq * 8;
    const unsigned short* bBase = w2t + (size_t)e * DMODEL * HDIM + (size_t)n0 * HDIM;
    unsigned short* sAdst = &sA[ar][aq * 8];
    const unsigned short* bPtr[4];
    unsigned short* sBdst[4];
#pragma unroll
    for (int rep = 0; rep < 4; rep++) {
        int lin = tid + rep * 256;
        int row = lin >> 2, q = lin & 3;
        bPtr[rep]  = bBase + (size_t)row * HDIM + q * 8;
        sBdst[rep] = &sB[row][q * 8];
    }

    v8f acc[2][4] = {};

    for (int k0 = 0; k0 < HDIM; k0 += 32) {
        __syncthreads();
        *(uint4*)sAdst = *(const uint4*)(aSrc + k0);
#pragma unroll
        for (int rep = 0; rep < 4; rep++)
            *(uint4*)sBdst[rep] = *(const uint4*)(bPtr[rep] + k0);
        __syncthreads();

        if (k0 + 32 < HDIM && aq == 0) {
            __builtin_prefetch(aSrc + k0 + 32, 0, 3);
#pragma unroll
            for (int rep = 0; rep < 4; rep++)
                __builtin_prefetch(bPtr[rep] + k0 + 32, 0, 3);
        }

        v16bf afrag[2], bfrag[4];
#pragma unroll
        for (int mi = 0; mi < 2; mi++) {
            int row = wm + mi * 16 + lm;
            v8bf a0 = *(const v8bf*)&sA[row][lg * 8];
            v8bf a1 = *(const v8bf*)&sA[row][16 + lg * 8];
            afrag[mi] = __builtin_shufflevector(a0, a1, 0,1,2,3,4,5,6,7,8,9,10,11,12,13,14,15);
        }
#pragma unroll
        for (int ni = 0; ni < 4; ni++) {
            int col = wn + ni * 16 + lm;
            v8bf b0 = *(const v8bf*)&sB[col][lg * 8];
            v8bf b1v = *(const v8bf*)&sB[col][16 + lg * 8];
            bfrag[ni] = __builtin_shufflevector(b0, b1v, 0,1,2,3,4,5,6,7,8,9,10,11,12,13,14,15);
        }
#pragma unroll
        for (int mi = 0; mi < 2; mi++)
#pragma unroll
            for (int ni = 0; ni < 4; ni++)
                acc[mi][ni] = __builtin_amdgcn_wmma_f32_16x16x32_bf16(
                    false, afrag[mi], false, bfrag[ni], (short)0, acc[mi][ni], false, false);
    }

    // epilogue: (acc + b2) * gate_weight, atomic-accumulate into out (= x residual)
    const float* bias = b2 + (size_t)e * DMODEL;
#pragma unroll
    for (int mi = 0; mi < 2; mi++)
#pragma unroll
        for (int ni = 0; ni < 4; ni++) {
            int col = n0 + wn + ni * 16 + lm;
            float bv = bias[col];
#pragma unroll
            for (int r = 0; r < 8; r++) {
                int row = wm + mi * 16 + r + lg * 8;
                int slotRow = base + mtile * 64 + row;
                int tok = map[slotRow];
                if (tok >= 0) {
                    float v = (acc[mi][ni][r] + bv) * wgt[slotRow];
                    atomicAdd(&out[(size_t)tok * DMODEL + col], v);
                }
            }
        }
}

// ---------------------------------------------------------------------------
extern "C" void kernel_launch(void* const* d_in, const int* in_sizes, int n_in,
                              void* d_out, int out_size, void* d_ws, size_t ws_size,
                              hipStream_t stream) {
    const float* x  = (const float*)d_in[0];
    const float* Wg = (const float*)d_in[1];
    const float* bg = (const float*)d_in[2];
    const float* W1 = (const float*)d_in[3];
    const float* b1 = (const float*)d_in[4];
    const float* W2 = (const float*)d_in[5];
    const float* b2 = (const float*)d_in[6];
    float* out = (float*)d_out;

    char* ws = (char*)d_ws;
    size_t off = 0;
#define WS_ALLOC(T, name, bytes) \
    T name = (T)(ws + off); off = (off + (size_t)(bytes) + 255) & ~(size_t)255;

    WS_ALLOC(unsigned short*, xb,  (size_t)N_TOK * DMODEL * 2)
    WS_ALLOC(unsigned short*, w1t, (size_t)NEXP * DMODEL * HDIM * 2)   // [E][H][D]
    WS_ALLOC(unsigned short*, w2t, (size_t)NEXP * HDIM * DMODEL * 2)   // [E][D][H]
    WS_ALLOC(unsigned short*, h1,  (size_t)SLOT_CAP * HDIM * 2)
    WS_ALLOC(int*,   map,    (size_t)SLOT_CAP * 4)
    WS_ALLOC(float*, wgt,    (size_t)SLOT_CAP * 4)
    WS_ALLOC(int*,   topi,   (size_t)N_TOK * TOPK * 4)
    WS_ALLOC(float*, topw,   (size_t)N_TOK * TOPK * 4)
    WS_ALLOC(int*,   counts, (size_t)NEXP * 4)
    WS_ALLOC(int*,   offp,   (size_t)(NEXP + 1) * 4)
    WS_ALLOC(int*,   cursor, (size_t)NEXP * 4)
#undef WS_ALLOC

    // 1) precision conversion; weights also transposed to N-major
    cvt_bf16_kernel<<<4096, 256, 0, stream>>>(x, xb, N_TOK * DMODEL);
    cvtT_kernel<<<dim3(HDIM / 32, DMODEL / 32, NEXP), dim3(32, 8), 0, stream>>>(
        W1, w1t, DMODEL, HDIM);
    cvtT_kernel<<<dim3(DMODEL / 32, HDIM / 32, NEXP), dim3(32, 8), 0, stream>>>(
        W2, w2t, HDIM, DMODEL);
    // 2) residual init + routing-state reset (every launch; graph-replay safe)
    init_kernel<<<4096, 256, 0, stream>>>(x, out, map, counts, cursor);
    // 3) gating + top-2
    gate_kernel<<<N_TOK / 8, 256, 0, stream>>>(x, Wg, bg, topi, topw, counts);
    // 4) padded per-expert offsets, slot scatter
    scan_kernel<<<1, 32, 0, stream>>>(counts, offp);
    scatter_kernel<<<N_TOK / 256, 256, 0, stream>>>(topi, topw, offp, cursor, map, wgt);
    // 5) expert FFN, routed: FC1 (+GELU) then FC2 (+gate weight, atomic combine)
    fc1_kernel<<<dim3(256, HDIM / 256, NEXP), 256, 0, stream>>>(xb, w1t, b1, map, offp, h1);
    fc2_kernel<<<dim3(256, DMODEL / 256, NEXP), 256, 0, stream>>>(h1, w2t, b2, map, offp, wgt, out);
}
